// Block_25477746000282
// MI455X (gfx1250) — matrix operand order repaired and checked by reading, not compiled, not run on hardware
//
#include <hip/hip_runtime.h>
#include <hip/hip_bf16.h>
#include <cmath>

#ifndef __has_builtin
#define __has_builtin(x) 0
#endif
#if __has_builtin(__builtin_amdgcn_tensor_load_to_lds)
#define HAVE_TDM 1
#else
#define HAVE_TDM 0
#endif

// ---------------------------------------------------------------------------
// Types
// ---------------------------------------------------------------------------
typedef __bf16 bf16;
typedef __attribute__((ext_vector_type(8)))  __bf16 v8bf;
typedef __attribute__((ext_vector_type(16))) __bf16 v16bf;
typedef __attribute__((ext_vector_type(8)))  float  v8f;
typedef __attribute__((ext_vector_type(4)))  unsigned u32x4;
typedef __attribute__((ext_vector_type(4)))  int      i32x4;
typedef __attribute__((ext_vector_type(8)))  int      i32x8;

#define D_MODEL   2048
#define N_HEADS   16
#define HD        128
#define SEQ       2048
#define BATCH     4
#define NROWS     (BATCH * SEQ)          // 8192
#define D_FF      (4 * D_MODEL)          // 8192
#define D_QKV     (3 * D_MODEL)          // 6144

__device__ __forceinline__ v16bf cat16(v8bf lo, v8bf hi) {
    v16bf r;
#pragma unroll
    for (int i = 0; i < 8; i++) { r[i] = lo[i]; r[i + 8] = hi[i]; }
    return r;
}
// A-fragment: lane L -> row m, elements [k0..k0+7] and [k0+16..k0+23]
__device__ __forceinline__ v16bf load_afrag(const bf16* p) {
    return cat16(*(const v8bf*)p, *(const v8bf*)(p + 16));
}
// B-fragment: lane L -> column n, K rows [k0..k0+7] and [k0+16..k0+23] (row stride LDB == AT_LD)
__device__ __forceinline__ v16bf load_bfrag(const bf16* p) {
    v16bf r;
#pragma unroll
    for (int i = 0; i < 8; i++) { r[i] = p[i * 136]; r[i + 8] = p[(16 + i) * 136]; }
    return r;
}
__device__ __forceinline__ v8f wmma_bf(v16bf a, v16bf b, v8f c) {
    return __builtin_amdgcn_wmma_f32_16x16x32_bf16(
        false, a, false, b, (short)0, c, false, false);
}

#if HAVE_TDM
// ---------------------------------------------------------------------------
// Tensor Data Mover: 2D tile load Global -> LDS (D# per ISA 08_async_tensor §8)
//   data_size = 2B; TDM writes the LDS row padding (bank skew) itself.
//   Issue from ONE wave; drain with s_wait_tensorcnt(0) + block barrier.
//   This toolchain (clang-23 / therock-10.0) uses the 6-arg builtin form.
// ---------------------------------------------------------------------------
__device__ __forceinline__ void tdm_load_2d_bf16(
    const void* gaddr, unsigned lds_byte_off,
    unsigned tile_d0, unsigned tile_d1,
    unsigned long long tensor_d0, unsigned long long tensor_d1,
    unsigned long long stride_elems,
    int pad_interval_code, int pad_amount_code)
{
    unsigned long long ga = (unsigned long long)gaddr;
    u32x4 g0;
    g0[0] = 1u;                                     // count=1 (user mode)
    g0[1] = lds_byte_off;                           // lds_addr
    g0[2] = (unsigned)(ga & 0xFFFFFFFFu);           // global_addr[31:0]
    g0[3] = (unsigned)((ga >> 32) & 0x1FFFFFFu)     // global_addr[56:32]
          | (2u << 30);                             // type = 2 (image)
    i32x8 g1;
    g1[0] = (1 << 16)                               // data_size = 2B
          | (1 << 20)                               // pad_enable
          | (pad_interval_code << 22)
          | (pad_amount_code << 25);
    g1[1] = (int)((tensor_d0 & 0xFFFFull) << 16);                       // dim0[15:0]
    g1[2] = (int)(((tensor_d0 >> 16) & 0xFFFFull)
          | ((tensor_d1 & 0xFFFFull) << 16));                           // dim0[31:16]|dim1[15:0]
    g1[3] = (int)(((tensor_d1 >> 16) & 0xFFFFull) | (tile_d0 << 16));   // dim1[31:16]|tile0
    g1[4] = (int)(tile_d1 & 0xFFFFu);                                   // tile1; tile2=0
    g1[5] = (int)(stride_elems & 0xFFFFFFFFull);                        // stride0[31:0]
    g1[6] = (int)((stride_elems >> 32) & 0xFFFFull);                    // stride0[47:32]
    g1[7] = 0;
    i32x4 z4 = {0, 0, 0, 0};
    i32x8 z8 = {0, 0, 0, 0, 0, 0, 0, 0};
    __builtin_amdgcn_tensor_load_to_lds(g0, g1, z4, z4, z8, 0);
}
#endif

// ---------------------------------------------------------------------------
// fp32 -> bf16 conversion
// ---------------------------------------------------------------------------
__global__ __launch_bounds__(256)
void cvt_kernel(const float* __restrict__ in, bf16* __restrict__ out, int n4) {
    int i = blockIdx.x * 256 + threadIdx.x;
    if (i < n4) {
        float4 v = ((const float4*)in)[i];
        union { bf16 h[4]; uint2 u; } pk;
        pk.h[0] = (bf16)v.x; pk.h[1] = (bf16)v.y;
        pk.h[2] = (bf16)v.z; pk.h[3] = (bf16)v.w;
        *(uint2*)(out + (size_t)i * 4) = pk.u;
    }
}

// ---------------------------------------------------------------------------
// LayerNorm: one row (2048) per 256-thread block, bf16 output
// ---------------------------------------------------------------------------
__global__ __launch_bounds__(256)
void ln_kernel(const float* __restrict__ x, const float* __restrict__ w,
               const float* __restrict__ b, bf16* __restrict__ out) {
    __shared__ float red[256];
    const int tid = threadIdx.x;
    const float* xr = x + (size_t)blockIdx.x * D_MODEL;

    float v[8];
    float4 a = ((const float4*)xr)[tid * 2 + 0];
    float4 c = ((const float4*)xr)[tid * 2 + 1];
    v[0] = a.x; v[1] = a.y; v[2] = a.z; v[3] = a.w;
    v[4] = c.x; v[5] = c.y; v[6] = c.z; v[7] = c.w;

    float s = 0.f;
#pragma unroll
    for (int i = 0; i < 8; i++) s += v[i];
    red[tid] = s; __syncthreads();
#pragma unroll
    for (int off = 128; off > 0; off >>= 1) {
        if (tid < off) red[tid] += red[tid + off];
        __syncthreads();
    }
    const float mu = red[0] * (1.0f / D_MODEL);
    __syncthreads();

    float s2 = 0.f;
#pragma unroll
    for (int i = 0; i < 8; i++) { float d = v[i] - mu; s2 += d * d; }
    red[tid] = s2; __syncthreads();
#pragma unroll
    for (int off = 128; off > 0; off >>= 1) {
        if (tid < off) red[tid] += red[tid + off];
        __syncthreads();
    }
    const float rstd = rsqrtf(red[0] * (1.0f / D_MODEL) + 1e-5f);

    bf16* orow = out + (size_t)blockIdx.x * D_MODEL;
#pragma unroll
    for (int i = 0; i < 8; i++) {
        int idx = tid * 8 + i;
        orow[idx] = (bf16)((v[i] - mu) * rstd * w[idx] + b[idx]);
    }
}

// ---------------------------------------------------------------------------
// Generic bf16 WMMA GEMM: C[M,N] = A[M,K] * W[N,K]^T + bias (+ epilogue)
//   mode 0: bf16(acc+bias)   mode 1: fp32(res+acc+bias)   mode 2: bf16(gelu)
// 256 threads (8 waves), 128x128 tile, BK=64.
// Double-buffered pipeline: TDM DMA of A(k+1) + global W(k+1) loads run
// under the WMMAs of tile k; W regs -> LDS (transposed) after compute.
// Buffer pointers are computed per-iteration (no pointer arrays -> no
// addrspacecast static initializers).
// ---------------------------------------------------------------------------
#define GBM 128
#define GBN 128
#define GBK 64
#define LDA 72     // GBK + 8  (144 B rows, 16B aligned, bank-skewed)
#define LDB 136    // GBN + 8  (272 B rows)
#define SA_BYTES (GBM * LDA * 2)
#define SB_BYTES (GBK * LDB * 2)

__global__ __launch_bounds__(256)
void gemm_bf16_kernel(const bf16* __restrict__ A, const bf16* __restrict__ W,
                      const float* __restrict__ bias, const float* __restrict__ res,
                      void* __restrict__ out, int M, int N, int K, int mode) {
    __shared__ __align__(16) char smem[2 * (SA_BYTES + SB_BYTES)];

    const int tid  = threadIdx.x;
    const int lane = tid & 31, wave = tid >> 5;
    const int wm = wave & 3, wn = wave >> 2;      // 4x2 wave grid
    const int m0 = blockIdx.y * GBM, n0 = blockIdx.x * GBN;
    const int mrow = lane & 15, hi = lane >> 4;
    const int np = (tid & 63) * 2, kcS = tid >> 6;   // W staging coords
    const int arow = tid & 127, accol = tid >> 7;    // A staging coords (fallback)

    v8f acc[2][4];
#pragma unroll
    for (int a = 0; a < 2; a++)
#pragma unroll
        for (int t = 0; t < 4; t++) acc[a][t] = (v8f)0.f;

    // ---- prologue: stage tile 0 into buffer 0 ----
#if HAVE_TDM
    if (wave == 0)
        tdm_load_2d_bf16(A + (size_t)m0 * K, /*lds_off=*/0,
                         GBK, GBM, (unsigned long long)K, (unsigned long long)M,
                         (unsigned long long)K, /*32 dwords=*/4, /*4 dwords=*/3);
#else
    {
        const bf16* g = A + (size_t)(m0 + arow) * K + accol * 32;
        bf16* s = (bf16*)smem + arow * LDA + accol * 32;
#pragma unroll
        for (int i = 0; i < 4; i++)
            *(uint4*)(s + i * 8) = *(const uint4*)(g + i * 8);
    }
#endif
    {
        bf16* sB0 = (bf16*)(smem + 2 * SA_BYTES);
        const bf16* g0 = W + (size_t)(n0 + np) * K + kcS * 16;
        const bf16* g1 = g0 + K;
#pragma unroll
        for (int c = 0; c < 2; c++) {
            v8bf a = *(const v8bf*)(g0 + c * 8);
            v8bf b = *(const v8bf*)(g1 + c * 8);
#pragma unroll
            for (int i = 0; i < 8; i++) {
                union { bf16 h[2]; unsigned u; } pr;
                pr.h[0] = a[i]; pr.h[1] = b[i];
                *(unsigned*)&sB0[(kcS * 16 + c * 8 + i) * LDB + np] = pr.u;
            }
        }
    }
#if HAVE_TDM
    if (wave == 0) __builtin_amdgcn_s_wait_tensorcnt((short)0);
#endif
    __syncthreads();

    const int nk = K / GBK;
    for (int kt = 0; kt < nk; kt++) {
        const int cur = kt & 1, nxt = cur ^ 1;
        const bool hasnext = (kt + 1 < nk);
        const int k1 = (kt + 1) * GBK;

        // ---- issue next tile: TDM for A, global->regs for W ----
        v8bf wr0, wr1, wr2, wr3;
#if !HAVE_TDM
        uint4 ar[4];
#endif
        if (hasnext) {
            const bf16* g0 = W + (size_t)(n0 + np) * K + k1 + kcS * 16;
            const bf16* g1 = g0 + K;
            wr0 = *(const v8bf*)g0;       wr1 = *(const v8bf*)(g0 + 8);
            wr2 = *(const v8bf*)g1;       wr3 = *(const v8bf*)(g1 + 8);
#if HAVE_TDM
            if (wave == 0)
                tdm_load_2d_bf16(A + (size_t)m0 * K + k1,
                                 (unsigned)(nxt * SA_BYTES),
                                 GBK, GBM, (unsigned long long)K,
                                 (unsigned long long)M, (unsigned long long)K, 4, 3);
#else
            const bf16* ga = A + (size_t)(m0 + arow) * K + k1 + accol * 32;
#pragma unroll
            for (int i = 0; i < 4; i++) ar[i] = *(const uint4*)(ga + i * 8);
#endif
            if (kt + 2 < nk) {                       // prefetch k+2 weights
                __builtin_prefetch(g0 + GBK, 0, 3);
                __builtin_prefetch(g1 + GBK, 0, 3);
            }
        }

        // ---- compute tile k from current buffers ----
        {
            const bf16* sA = (const bf16*)(smem + cur * SA_BYTES);
            const bf16* sB = (const bf16*)(smem + 2 * SA_BYTES + cur * SB_BYTES);
#pragma unroll
            for (int kc = 0; kc < 2; kc++) {
                const bf16* ap = &sA[(wm * 32 + mrow) * LDA + kc * 32 + hi * 8];
                v16bf a0 = load_afrag(ap);
                v16bf a1 = load_afrag(ap + 16 * LDA);
                v16bf bfr[4];
#pragma unroll
                for (int nt = 0; nt < 4; nt++)
                    bfr[nt] = load_bfrag(&sB[(kc * 32 + hi * 8) * LDB + wn * 64 + nt * 16 + mrow]);
#pragma unroll
                for (int nt = 0; nt < 4; nt++) {
                    acc[0][nt] = wmma_bf(a0, bfr[nt], acc[0][nt]);
                    acc[1][nt] = wmma_bf(a1, bfr[nt], acc[1][nt]);
                }
            }
        }

        // ---- publish next tile into alternate buffers ----
        if (hasnext) {
            bf16* sBn = (bf16*)(smem + 2 * SA_BYTES + nxt * SB_BYTES);
#pragma unroll
            for (int i = 0; i < 8; i++) {
                union { bf16 h[2]; unsigned u; } pr;
                pr.h[0] = wr0[i]; pr.h[1] = wr2[i];
                *(unsigned*)&sBn[(kcS * 16 + i) * LDB + np] = pr.u;
            }
#pragma unroll
            for (int i = 0; i < 8; i++) {
                union { bf16 h[2]; unsigned u; } pr;
                pr.h[0] = wr1[i]; pr.h[1] = wr3[i];
                *(unsigned*)&sBn[(kcS * 16 + 8 + i) * LDB + np] = pr.u;
            }
#if !HAVE_TDM
            bf16* sAn = (bf16*)(smem + nxt * SA_BYTES);
#pragma unroll
            for (int i = 0; i < 4; i++)
                *(uint4*)&sAn[arow * LDA + accol * 32 + i * 8] = ar[i];
#endif
        }
#if HAVE_TDM
        if (hasnext && wave == 0) __builtin_amdgcn_s_wait_tensorcnt((short)0);
#endif
        __syncthreads();
    }

    // ---- epilogue ----
#pragma unroll
    for (int mi = 0; mi < 2; mi++) {
#pragma unroll
        for (int nt = 0; nt < 4; nt++) {
            const int col = n0 + wn * 64 + nt * 16 + (lane & 15);
            const float bv = bias[col];
#pragma unroll
            for (int r = 0; r < 8; r++) {
                const int row = m0 + wm * 32 + mi * 16 + hi * 8 + r;
                const size_t idx = (size_t)row * N + col;
                float v = acc[mi][nt][r] + bv;
                if (mode == 0) {
                    ((bf16*)out)[idx] = (bf16)v;
                } else if (mode == 1) {
                    ((float*)out)[idx] = res[idx] + v;
                } else {
                    float g = 0.5f * v * (1.0f + erff(v * 0.70710678118654752f));
                    ((bf16*)out)[idx] = (bf16)g;
                }
            }
        }
    }
}

// ---------------------------------------------------------------------------
// Flash attention with ALiBi + causal mask.
// 256 threads (8 waves); one (b, h, 128-q-rows) per block.
// V tile DMA'd by TDM (overlaps with manual K^T transpose staging).
// ---------------------------------------------------------------------------
#define AT_LD 136                       // 128 + 8 pad (272 B rows)
#define AT_TILE (128 * AT_LD * 2)       // 34816 B

__global__ __launch_bounds__(256)
void attn_kernel(const bf16* __restrict__ qkv, const float* __restrict__ slopes,
                 bf16* __restrict__ out) {
    __shared__ __align__(16) char smem[3 * AT_TILE];
    bf16* sKt = (bf16*)smem;                       // K^T (d,j); Q staging (off 0)
    bf16* sV  = (bf16*)(smem + AT_TILE);           // V (j,d)    (off AT_TILE)
    bf16* sP  = (bf16*)(smem + 2 * AT_TILE);       // P, 16 rows per wave

    const int tid  = threadIdx.x;
    const int lane = tid & 31, wave = tid >> 5;
    const int mrow = lane & 15, hi = lane >> 4;
    const int qb = blockIdx.x;
    const int b  = blockIdx.y >> 4;
    const int h  = blockIdx.y & 15;
    const float slope = slopes[h];
    const float scale = 0.08838834764831845f;      // 1/sqrt(128)

    const size_t rs = D_QKV;
    const bf16* Qg  = qkv + (size_t)(b * SEQ + qb * 128) * rs + h * HD;
    const bf16* KVb = qkv + (size_t)(b * SEQ) * rs + h * HD;

    // ---- stage Q (row-major) into sKt ----
#if HAVE_TDM
    if (wave == 0) {
        tdm_load_2d_bf16(Qg, /*lds_off=*/0, 128, 128,
                         (unsigned long long)rs, (unsigned long long)NROWS,
                         (unsigned long long)rs,
                         /*64 dwords=*/5, /*4 dwords=*/3);
        __builtin_amdgcn_s_wait_tensorcnt((short)0);
    }
#else
    {
        const int r = tid & 127, dc = tid >> 7;
        const bf16* g = Qg + (size_t)r * rs + dc * 64;
        bf16* s = &sKt[r * AT_LD + dc * 64];
#pragma unroll
        for (int i = 0; i < 8; i++)
            *(uint4*)(s + i * 8) = *(const uint4*)(g + i * 8);
    }
#endif
    __syncthreads();
    v16bf qf[4];
#pragma unroll
    for (int kc = 0; kc < 4; kc++)
        qf[kc] = load_afrag(&sKt[(wave * 16 + mrow) * AT_LD + kc * 32 + hi * 8]);
    __syncthreads();

    v8f O[8];
#pragma unroll
    for (int nt = 0; nt < 8; nt++) O[nt] = (v8f)0.f;
    float m_i[8], l_i[8];
#pragma unroll
    for (int r = 0; r < 8; r++) { m_i[r] = -INFINITY; l_i[r] = 0.f; }

    const int iq0 = qb * 128 + wave * 16 + hi * 8;

    for (int kb = 0; kb <= qb; kb++) {
        __syncthreads();
        // ---- stage V via TDM (overlaps with manual K^T staging below) ----
#if HAVE_TDM
        if (wave == 0) {
            tdm_load_2d_bf16(KVb + (size_t)(kb * 128) * rs + 2 * D_MODEL,
                             /*lds_off=*/AT_TILE, 128, 128,
                             (unsigned long long)rs, (unsigned long long)NROWS,
                             (unsigned long long)rs, 5, 3);
        }
#else
        {
            const int r = tid & 127, dc = tid >> 7;
            const bf16* g = KVb + (size_t)(kb * 128 + r) * rs + 2 * D_MODEL + dc * 64;
            bf16* s = &sV[r * AT_LD + dc * 64];
#pragma unroll
            for (int i = 0; i < 8; i++)
                *(uint4*)(s + i * 8) = *(const uint4*)(g + i * 8);
        }
#endif
        // ---- stage K^T manually (TDM cannot transpose): sKt[d][j] ----
        {
            const int jp = (tid & 63) * 2, dc = tid >> 6;
            const bf16* g0 = KVb + (size_t)(kb * 128 + jp) * rs + D_MODEL + dc * 32;
            const bf16* g1 = g0 + rs;
#pragma unroll
            for (int c = 0; c < 4; c++) {
                v8bf a  = *(const v8bf*)(g0 + c * 8);
                v8bf bb = *(const v8bf*)(g1 + c * 8);
#pragma unroll
                for (int i = 0; i < 8; i++) {
                    int d = dc * 32 + c * 8 + i;
                    union { bf16 h[2]; unsigned u; } pr;
                    pr.h[0] = a[i]; pr.h[1] = bb[i];
                    *(unsigned*)&sKt[d * AT_LD + jp] = pr.u;
                }
            }
        }
#if HAVE_TDM
        if (wave == 0) __builtin_amdgcn_s_wait_tensorcnt((short)0);
#endif
        __syncthreads();

        // ---- scores: 8 tiles, processed as independent pairs for ILP ----
        v8f st[8];
#pragma unroll
        for (int np = 0; np < 4; np++) {
            v8f c0 = (v8f)0.f, c1 = (v8f)0.f;
#pragma unroll
            for (int kc = 0; kc < 4; kc++) {
                const bf16* kr = &sKt[(kc * 32 + hi * 8) * AT_LD + mrow];
                v16bf k0f = load_bfrag(kr + (2 * np + 0) * 16);
                v16bf k1f = load_bfrag(kr + (2 * np + 1) * 16);
                c0 = wmma_bf(qf[kc], k0f, c0);
                c1 = wmma_bf(qf[kc], k1f, c1);
            }
            st[2 * np + 0] = c0;
            st[2 * np + 1] = c1;
        }

        // ---- scale + ALiBi + causal mask; row max ----
        float rowmax[8];
#pragma unroll
        for (int r = 0; r < 8; r++) rowmax[r] = -INFINITY;
#pragma unroll
        for (int nt = 0; nt < 8; nt++) {
            const int j = kb * 128 + nt * 16 + (lane & 15);
#pragma unroll
            for (int r = 0; r < 8; r++) {
                const int i = iq0 + r;
                float v = st[nt][r] * scale + slope * (float)(j - i);
                v = (j <= i) ? v : -INFINITY;
                st[nt][r] = v;
                rowmax[r] = fmaxf(rowmax[r], v);
            }
        }
#pragma unroll
        for (int r = 0; r < 8; r++)
#pragma unroll
            for (int msk = 1; msk < 16; msk <<= 1)
                rowmax[r] = fmaxf(rowmax[r], __shfl_xor(rowmax[r], msk, 16));

        // ---- online softmax ----
        float fac[8], rsum[8];
#pragma unroll
        for (int r = 0; r < 8; r++) {
            float mn = fmaxf(m_i[r], rowmax[r]);
            fac[r] = __expf(m_i[r] - mn);
            m_i[r] = mn;
            rsum[r] = 0.f;
        }
#pragma unroll
        for (int nt = 0; nt < 8; nt++)
#pragma unroll
            for (int r = 0; r < 8; r++) {
                float p = __expf(st[nt][r] - m_i[r]);
                st[nt][r] = p;
                rsum[r] += p;
            }
#pragma unroll
        for (int r = 0; r < 8; r++)
#pragma unroll
            for (int msk = 1; msk < 16; msk <<= 1)
                rsum[r] += __shfl_xor(rsum[r], msk, 16);
#pragma unroll
        for (int r = 0; r < 8; r++) l_i[r] = l_i[r] * fac[r] + rsum[r];
#pragma unroll
        for (int nt = 0; nt < 8; nt++)
#pragma unroll
            for (int r = 0; r < 8; r++) O[nt][r] *= fac[r];

        // ---- P (C-layout) -> per-wave LDS (A-layout source) ----
#pragma unroll
        for (int nt = 0; nt < 8; nt++)
#pragma unroll
            for (int r = 0; r < 8; r++)
                sP[(wave * 16 + hi * 8 + r) * AT_LD + nt * 16 + (lane & 15)] =
                    (bf16)st[nt][r];

        // ---- O += P * V ----
#pragma unroll
        for (int kc = 0; kc < 4; kc++) {
            v16bf pf = load_afrag(&sP[(wave * 16 + mrow) * AT_LD + kc * 32 + hi * 8]);
#pragma unroll
            for (int nt = 0; nt < 8; nt++) {
                v16bf vf = load_bfrag(&sV[(kc * 32 + hi * 8) * AT_LD + nt * 16 + mrow]);
                O[nt] = wmma_bf(pf, vf, O[nt]);
            }
        }
    }

    // ---- normalize and store ----
    float invl[8];
#pragma unroll
    for (int r = 0; r < 8; r++) invl[r] = 1.0f / l_i[r];
#pragma unroll
    for (int nt = 0; nt < 8; nt++)
#pragma unroll
        for (int r = 0; r < 8; r++) {
            const int i = iq0 + r;
            const int d = nt * 16 + (lane & 15);
            out[(size_t)(b * SEQ + i) * D_MODEL + h * HD + d] =
                (bf16)(O[nt][r] * invl[r]);
        }
}

// ---------------------------------------------------------------------------
// Host launcher
// ---------------------------------------------------------------------------
extern "C" void kernel_launch(void* const* d_in, const int* in_sizes, int n_in,
                              void* d_out, int out_size, void* d_ws, size_t ws_size,
                              hipStream_t stream) {
    const float* x     = (const float*)d_in[0];
    const float* ln1w  = (const float*)d_in[1];
    const float* ln1b  = (const float*)d_in[2];
    const float* Wqkv  = (const float*)d_in[3];
    const float* bqkv  = (const float*)d_in[4];
    const float* Wo    = (const float*)d_in[5];
    const float* bo    = (const float*)d_in[6];
    const float* ln2w  = (const float*)d_in[7];
    const float* ln2b  = (const float*)d_in[8];
    const float* W1    = (const float*)d_in[9];
    const float* b1    = (const float*)d_in[10];
    const float* W2    = (const float*)d_in[11];
    const float* b2    = (const float*)d_in[12];
    const float* slope = (const float*)d_in[13];

    char* p = (char*)d_ws;
    bf16*  hln   = (bf16*)p;  p += (size_t)NROWS * D_MODEL * 2;
    bf16*  qkvb  = (bf16*)p;  p += (size_t)NROWS * D_QKV  * 2;
    bf16*  attnb = (bf16*)p;  p += (size_t)NROWS * D_MODEL * 2;
    float* x2    = (float*)p; p += (size_t)NROWS * D_MODEL * 4;
    bf16*  hln2  = (bf16*)p;  p += (size_t)NROWS * D_MODEL * 2;
    bf16*  h1b   = (bf16*)p;  p += (size_t)NROWS * D_FF   * 2;
    bf16*  Wqkvb = (bf16*)p;  p += (size_t)D_QKV * D_MODEL * 2;
    bf16*  Wob   = (bf16*)p;  p += (size_t)D_MODEL * D_MODEL * 2;
    bf16*  W1b   = (bf16*)p;  p += (size_t)D_FF * D_MODEL * 2;
    bf16*  W2b   = (bf16*)p;  p += (size_t)D_MODEL * D_FF * 2;

    int n4;
    n4 = (D_QKV * D_MODEL) / 4;
    cvt_kernel<<<(n4 + 255) / 256, 256, 0, stream>>>(Wqkv, Wqkvb, n4);
    n4 = (D_MODEL * D_MODEL) / 4;
    cvt_kernel<<<(n4 + 255) / 256, 256, 0, stream>>>(Wo, Wob, n4);
    n4 = (D_FF * D_MODEL) / 4;
    cvt_kernel<<<(n4 + 255) / 256, 256, 0, stream>>>(W1, W1b, n4);
    n4 = (D_MODEL * D_FF) / 4;
    cvt_kernel<<<(n4 + 255) / 256, 256, 0, stream>>>(W2, W2b, n4);

    ln_kernel<<<NROWS, 256, 0, stream>>>(x, ln1w, ln1b, hln);

    gemm_bf16_kernel<<<dim3(D_QKV / GBN, NROWS / GBM), 256, 0, stream>>>(
        hln, Wqkvb, bqkv, nullptr, qkvb, NROWS, D_QKV, D_MODEL, 0);

    attn_kernel<<<dim3(SEQ / 128, BATCH * N_HEADS), 256, 0, stream>>>(
        qkvb, slope, attnb);

    gemm_bf16_kernel<<<dim3(D_MODEL / GBN, NROWS / GBM), 256, 0, stream>>>(
        attnb, Wob, bo, x, x2, NROWS, D_MODEL, D_MODEL, 1);

    ln_kernel<<<NROWS, 256, 0, stream>>>(x2, ln2w, ln2b, hln2);

    gemm_bf16_kernel<<<dim3(D_FF / GBN, NROWS / GBM), 256, 0, stream>>>(
        hln2, W1b, b1, nullptr, h1b, NROWS, D_FF, D_MODEL, 2);

    gemm_bf16_kernel<<<dim3(D_MODEL / GBN, NROWS / GBM), 256, 0, stream>>>(
        h1b, W2b, b2, x2, (float*)d_out, NROWS, D_MODEL, D_FF, 1);
}
